// STDP_10599979287276
// MI455X (gfx1250) — matrix operand order repaired and checked by reading, not compiled, not run on hardware
//
#include <hip/hip_runtime.h>

typedef __attribute__((ext_vector_type(16))) _Float16 v16h;
typedef __attribute__((ext_vector_type(8)))  _Float16 v8h;
typedef __attribute__((ext_vector_type(8)))  float    v8f;

namespace {
constexpr int kT       = 2048;  // timesteps
constexpr int kB       = 128;   // batch
constexpr int kNin     = 256;   // pre-synaptic neurons  (delta_w columns)
constexpr int kNout    = 128;   // post-synaptic neurons (delta_w rows)
constexpr int kCT      = 32;    // timesteps per chunk
constexpr int kHalo    = 32;    // trace warm-up steps: 0.5^32 ~ 2e-10 -> negligible
constexpr int kBG      = 32;    // batch rows per workgroup == WMMA K per step
constexpr int kThreads = 512;   // 16 wave32 waves
constexpr int kRS      = 40;    // staging row stride in halves (80B: 16B-aligned, bank-skewed)
}

__global__ void __launch_bounds__(kThreads)
stdp_zero(float* p) {
  p[blockIdx.x * kThreads + threadIdx.x] = 0.0f;
}

__global__ void __launch_bounds__(kThreads)
stdp_dw(const float* __restrict__ in_sp,   // [T, B, NIN]  binary spikes (0/1)
        const float* __restrict__ out_sp,  // [T, B, NOUT] binary spikes (0/1)
        float* __restrict__ dw)            // [NOUT, NIN]
{
  // Double-buffered f16 fragment staging (spikes are 0/1 -> f16 is lossless).
  __shared__ alignas(16) _Float16 trH[2][kNin * kRS];   // 2 x 20 KB, [i][b]: B fragments
  __shared__ alignas(16) _Float16 soH[2][kNout * kRS];  // 2 x 10 KB, [o][b]: A fragments

  const int tid    = threadIdx.x;
  const int bgid   = blockIdx.x & 3;   // batch group (K split)
  const int tcid   = blockIdx.x >> 2;  // time chunk
  const int t0     = tcid * kCT;
  const int bstart = bgid * kBG;

  // Trace ownership: column i_own, batches [bh*16, bh*16+16).
  // Lanes vary i_own -> each of the 16 dword loads below is one 128B line.
  const int i_own = tid & (kNin - 1);
  const int bh    = tid >> 8;
  // A-staging ownership: column o_own, batches [q*8, q*8+8); coalesced likewise.
  const int o_own = tid & (kNout - 1);
  const int q     = tid >> 7;

  float tr[16];
#pragma unroll
  for (int j = 0; j < 16; ++j) tr[j] = 0.0f;

  float gin[16];   // s_in[t][bh*16+j][i_own], one step ahead
  float gout[8];   // s_out[t][q*8+j][o_own], one step ahead

  auto load_in = [&](int t) {
    const float* g = in_sp + ((size_t)t * kB + bstart + bh * 16) * kNin + i_own;
#pragma unroll
    for (int j = 0; j < 16; ++j) gin[j] = g[j * kNin];
  };
  auto load_out = [&](int t) {
    const float* g = out_sp + ((size_t)t * kB + bstart + q * 8) * kNout + o_own;
#pragma unroll
    for (int j = 0; j < 8; ++j) gout[j] = g[j * kNout];
  };
  auto update_trace = [&]() {
#pragma unroll
    for (int j = 0; j < 16; ++j)
      tr[j] = tr[j] * 0.5f + gin[j];  // trace - trace/tau + spike, tau=2 (bit-exact)
  };
  auto stage = [&](int buf) {
    // B fragments: trace (K=b) x (N=i); row i holds halves b=0..31.
    v8h h0, h1;
#pragma unroll
    for (int j = 0; j < 8; ++j) { h0[j] = (_Float16)tr[j]; h1[j] = (_Float16)tr[j + 8]; }
    *(v8h*)&trH[buf][i_own * kRS + bh * 16]     = h0;
    *(v8h*)&trH[buf][i_own * kRS + bh * 16 + 8] = h1;
    // A fragments: s_out^T (M=o) x (K=b); row o holds halves b=0..31.
    v8h ho;
#pragma unroll
    for (int j = 0; j < 8; ++j) ho[j] = (_Float16)gout[j];
    *(v8h*)&soH[buf][o_own * kRS + q * 8] = ho;
  };

  // WMMA tiling: 8 M-tiles (o) x 16 N-tiles (i) = 128 C-tiles over 16 waves.
  const int lane = tid & 31;
  const int wv   = tid >> 5;
  const int mt   = wv >> 1;        // M-tile 0..7
  const int nbas = (wv & 1) * 8;   // N-tiles nbas..nbas+7
  const int nloc = lane & 15;
  const int hi   = lane >> 4;

  v8f acc[8];
#pragma unroll
  for (int j = 0; j < 8; ++j)
#pragma unroll
    for (int r = 0; r < 8; ++r) acc[j][r] = 0.0f;

  auto do_wmma = [&](int buf) {
    // A 16x32 f16, contiguous k order chosen here (same as B): lane m=nloc, halves h -> K=hi*16 + h
    const int orow = mt * 16 + nloc;
    v8h alo = *(const v8h*)&soH[buf][orow * kRS + hi * 16];
    v8h ahi = *(const v8h*)&soH[buf][orow * kRS + hi * 16 + 8];
    v16h a = __builtin_shufflevector(alo, ahi, 0,1,2,3,4,5,6,7,8,9,10,11,12,13,14,15);
    // Preload ALL 8 B fragments; the sched_barrier below pins this grouping so
    // the 16 ds_load_b128 overlap each other, then one s_wait_dscnt, then
    // 8 back-to-back WMMAs (distinct D registers: no hazards, no exposed waits).
    v16h bf[8];
#pragma unroll
    for (int j = 0; j < 8; ++j) {
      const int irow = (nbas + j) * 16 + nloc;  // B 32x16, same contiguous k order as A: lane n=nloc, halves -> K=hi*16+h
      v8h blo = *(const v8h*)&trH[buf][irow * kRS + hi * 16];
      v8h bhi = *(const v8h*)&trH[buf][irow * kRS + hi * 16 + 8];
      bf[j] = __builtin_shufflevector(blo, bhi, 0,1,2,3,4,5,6,7,8,9,10,11,12,13,14,15);
    }
    __builtin_amdgcn_sched_barrier(0);  // keep loads above, WMMAs below
#pragma unroll
    for (int j = 0; j < 8; ++j)
      acc[j] = __builtin_amdgcn_wmma_f32_16x16x32_f16(
          false, a, false, bf[j], (short)0, acc[j], false, false);
  };

  // ---- halo: warm the pre-synaptic trace. Pure register recurrence, no LDS,
  // ---- no barriers; one-step-ahead global prefetch.
  int th0 = t0 - kHalo; if (th0 < 0) th0 = 0;
  load_in(th0);
  for (int t = th0; t < t0; ++t) {
    update_trace();      // consumes gin(t)
    load_in(t + 1);      // t+1 <= t0, always in range
  }

  // ---- main: one barrier per timestep with double-buffered fragments.
  // stage(k+1) writes buf^1 while other waves' do_wmma(k) still reads buf.
  load_out(t0);
  int buf = 0;
  for (int t = t0; t < t0 + kCT; ++t) {
    update_trace();
    stage(buf);
    if (t + 1 < t0 + kCT) { load_in(t + 1); load_out(t + 1); }
    __syncthreads();
    do_wmma(buf);
    buf ^= 1;
  }

  // ---- fold partial delta_w into global output ----
  // C 16x16 f32 layout: lane n=nloc, reg r -> M = hi*8 + r.
#pragma unroll
  for (int j = 0; j < 8; ++j) {
    const int icol = (nbas + j) * 16 + nloc;
#pragma unroll
    for (int r = 0; r < 8; ++r) {
      const int orow = mt * 16 + hi * 8 + r;
      unsafeAtomicAdd(&dw[orow * kNin + icol], acc[j][r]);
    }
  }
}

extern "C" void kernel_launch(void* const* d_in, const int* in_sizes, int n_in,
                              void* d_out, int out_size, void* d_ws, size_t ws_size,
                              hipStream_t stream) {
  (void)in_sizes; (void)n_in; (void)out_size; (void)d_ws; (void)ws_size;
  const float* in_sp  = (const float*)d_in[0];   // [2048,128,256] f32
  const float* out_sp = (const float*)d_in[1];   // [2048,128,128] f32
  float* dw = (float*)d_out;                     // [128,256] f32

  stdp_zero<<<(kNout * kNin) / kThreads, kThreads, 0, stream>>>(dw);
  stdp_dw<<<(kT / kCT) * (kB / kBG), kThreads, 0, stream>>>(in_sp, out_sp, dw);
}
